// Attention_80857054315218
// MI455X (gfx1250) — matrix operand order repaired and checked by reading, not compiled, not run on hardware
//
#include <hip/hip_runtime.h>
#include <hip/hip_bf16.h>

typedef __bf16 bf16_t;
typedef __attribute__((ext_vector_type(16))) __bf16 bf16x16;
typedef __attribute__((ext_vector_type(8)))  __bf16 bf16x8;
typedef __attribute__((ext_vector_type(8)))  float  f32x8;

#define HEADS   24
#define DHEAD   128
#define SEQ_IMG 2048
#define SEQ_TXT 512
#define SEQ_ALL 2560
#define DMODEL  3072

// ---------------------------------------------------------------- helpers --

// Hardware float->bf16 convert (backend lowers fptrunc to v_cvt_pk_bf16_f32).
__device__ __forceinline__ bf16_t f2bf(float f) { return (bf16_t)f; }

__device__ __forceinline__ f32x8 zero8() {
  f32x8 z;
  #pragma unroll
  for (int i = 0; i < 8; ++i) z[i] = 0.0f;
  return z;
}

__device__ __forceinline__ bf16x16 join8(bf16x8 lo, bf16x8 hi) {
  return __builtin_shufflevector(lo, hi, 0, 1, 2, 3, 4, 5, 6, 7,
                                         8, 9, 10, 11, 12, 13, 14, 15);
}

// A-fragment (16x32 bf16), source row-major with leading dim `ld`:
// lane L: row m = L%16; lanes 0-15 hold K 0-7 / 16-23, lanes 16-31 K 8-15 / 24-31.
__device__ __forceinline__ bf16x16 load_a_frag(const bf16_t* p, int ld) {
  int lane = threadIdx.x & 31;
  int m  = lane & 15;
  int kb = (lane >> 4) * 8;
  bf16x8 lo = *(const bf16x8*)(p + m * ld + kb);
  bf16x8 hi = *(const bf16x8*)(p + m * ld + kb + 16);
  return join8(lo, hi);
}

// B-fragment (32x16 bf16) from K-contiguous storage: element (k,n) = p[n*ld + k].
__device__ __forceinline__ bf16x16 load_bT_frag(const bf16_t* p, int ld) {
  int lane = threadIdx.x & 31;
  int n  = lane & 15;
  int kb = (lane >> 4) * 8;
  bf16x8 lo = *(const bf16x8*)(p + n * ld + kb);
  bf16x8 hi = *(const bf16x8*)(p + n * ld + kb + 16);
  return join8(lo, hi);
}

// CDNA5 async global->LDS copy (ASYNCcnt-tracked), 16 bytes per lane.
__device__ __forceinline__ void async_ld_b128(bf16_t* lds, const bf16_t* g) {
  unsigned l = (unsigned)(size_t)lds;   // LDS aperture: low 32 bits = LDS offset
  asm volatile("global_load_async_to_lds_b128 %0, %1, off"
               :: "v"(l), "v"(g) : "memory");
}
__device__ __forceinline__ void wait_async0() {
  asm volatile("s_wait_asynccnt 0" ::: "memory");
}

#define WMMA_BF16(A, B, C) \
  __builtin_amdgcn_wmma_f32_16x16x32_bf16(false, (A), false, (B), (short)0, (C), false, false)

// ------------------------------------------------------------- GEMM kernel --
// C[M,N] = A[M,K] * B[K,N] (+ bias[N]); A,B fp32 in memory, bf16 WMMA math.
// (fp32 sources must be converted while staging, so this path stays synchronous
//  with v_cvt_pk_bf16_f32; the bf16-native attention kernel uses async-to-LDS.)
__global__ __launch_bounds__(256)
void k_gemm_bf16(const float* __restrict__ A, const float* __restrict__ B,
                 const float* __restrict__ bias, float* __restrict__ C,
                 int M, int K, int N) {
  __shared__ alignas(16) bf16_t sA[128 * 32];    // [row][k]
  __shared__ alignas(16) bf16_t sBt[128 * 32];   // [col][k]  (transposed B tile)

  int tid  = threadIdx.x;
  int lane = tid & 31;
  int wave = tid >> 5;
  int tileM = blockIdx.y * 128;
  int tileN = blockIdx.x * 128;
  int mw = (wave & 3) * 32;
  int nw = (wave >> 2) * 64;

  int arow = tid >> 1;            // A: one 16-elem row segment
  int acol = (tid & 1) * 16;
  int bcol = tid & 127;           // B: one 16-elem column segment
  int bkg  = (tid >> 7) * 16;

  f32x8 acc[2][4];
  #pragma unroll
  for (int i = 0; i < 2; ++i)
    #pragma unroll
    for (int j = 0; j < 4; ++j) acc[i][j] = zero8();

  for (int k0 = 0; k0 < K; k0 += 32) {
    // ---- issue ALL global loads first (batched, latency overlapped) ----
    float4 av[4];
    #pragma unroll
    for (int it = 0; it < 4; ++it)
      av[it] = *(const float4*)(A + (size_t)(tileM + arow) * K + k0 + acol + it * 4);
    float bv[16];
    #pragma unroll
    for (int j = 0; j < 16; ++j)
      bv[j] = B[(size_t)(k0 + bkg + j) * N + tileN + bcol];

    if (k0 + 32 < K) {   // global_prefetch_b8 for next K-tile
      __builtin_prefetch((const void*)(A + (size_t)(tileM + arow) * K + k0 + 32 + acol), 0, 1);
      __builtin_prefetch((const void*)(B + (size_t)(k0 + 32 + bkg) * N + tileN + bcol), 0, 1);
    }

    // ---- convert fp32->bf16 (v_cvt_pk_bf16_f32) and store 32B runs ----
    #pragma unroll
    for (int it = 0; it < 4; ++it) {
      sA[arow * 32 + acol + it * 4 + 0] = f2bf(av[it].x);
      sA[arow * 32 + acol + it * 4 + 1] = f2bf(av[it].y);
      sA[arow * 32 + acol + it * 4 + 2] = f2bf(av[it].z);
      sA[arow * 32 + acol + it * 4 + 3] = f2bf(av[it].w);
    }
    #pragma unroll
    for (int j = 0; j < 16; ++j)
      sBt[bcol * 32 + bkg + j] = f2bf(bv[j]);

    __syncthreads();

    bf16x16 a0 = load_a_frag(&sA[(mw +  0) * 32], 32);
    bf16x16 a1 = load_a_frag(&sA[(mw + 16) * 32], 32);
    #pragma unroll
    for (int nt = 0; nt < 4; ++nt) {
      bf16x16 b = load_bT_frag(&sBt[(nw + nt * 16) * 32], 32);
      acc[0][nt] = WMMA_BF16(a0, b, acc[0][nt]);
      acc[1][nt] = WMMA_BF16(a1, b, acc[1][nt]);
    }
    __syncthreads();
  }

  int n  = lane & 15;
  int mb = (lane >> 4) * 8;
  #pragma unroll
  for (int mi = 0; mi < 2; ++mi)
    #pragma unroll
    for (int nt = 0; nt < 4; ++nt) {
      int col = tileN + nw + nt * 16 + n;
      float badd = bias ? bias[col] : 0.0f;
      #pragma unroll
      for (int r = 0; r < 8; ++r) {
        int row = tileM + mw + mi * 16 + mb + r;
        C[(size_t)row * N + col] = acc[mi][nt][r] + badd;
      }
    }
}

// ------------------------------------------------------ RMSNorm + RoPE ------
__device__ __forceinline__ void norm_rope_store(const float* __restrict__ src,
                                                const float* __restrict__ g,
                                                const float* __restrict__ fc,
                                                int s, int lane, float scale,
                                                bf16_t* __restrict__ dst) {
  int d = lane * 4;
  float4 x = *(const float4*)(src + d);
  float ss = x.x*x.x + x.y*x.y + x.z*x.z + x.w*x.w;
  #pragma unroll
  for (int m = 1; m < 32; m <<= 1) ss += __shfl_xor(ss, m, 32);
  float inv = rsqrtf(ss * (1.0f / 128.0f) + 1e-5f);
  float y0 = x.x * inv * g[d + 0];
  float y1 = x.y * inv * g[d + 1];
  float y2 = x.z * inv * g[d + 2];
  float y3 = x.w * inv * g[d + 3];
  const float* f = fc + (size_t)s * 128 + d;   // (cos,sin) pairs
  float c0 = f[0], s0 = f[1], c1 = f[2], s1 = f[3];
  dst[d + 0] = f2bf((y0 * c0 - y1 * s0) * scale);
  dst[d + 1] = f2bf((y0 * s0 + y1 * c0) * scale);
  dst[d + 2] = f2bf((y2 * c1 - y3 * s1) * scale);
  dst[d + 3] = f2bf((y2 * s1 + y3 * c1) * scale);
}

// q,k packed [h][s][d]; v packed TRANSPOSED [h][d][s] so attention V tiles
// are K-contiguous for both async staging and fragment loads.
__global__ __launch_bounds__(256)
void k_rmsrope(const float* __restrict__ qf, const float* __restrict__ kf,
               const float* __restrict__ vf,
               const float* __restrict__ gq, const float* __restrict__ gk,
               const float* __restrict__ fc,
               bf16_t* __restrict__ qb, bf16_t* __restrict__ kb,
               bf16_t* __restrict__ vT, int soff, int slocal) {
  int wave = threadIdx.x >> 5;
  int lane = threadIdx.x & 31;
  int row  = blockIdx.x * 8 + wave;
  int s = row / HEADS;
  int h = row % HEADS;
  if (s >= slocal) return;

  const float qscale = 0.08838834764831845f;   // 1/sqrt(128)
  size_t srcoff = (size_t)s * DMODEL + h * DHEAD;
  size_t dstoff = ((size_t)h * SEQ_ALL + soff + s) * DHEAD;

  norm_rope_store(qf + srcoff, gq, fc, s, lane, qscale, qb + dstoff);
  norm_rope_store(kf + srcoff, gk, fc, s, lane, 1.0f,   kb + dstoff);

  int d = lane * 4;
  float4 xv = *(const float4*)(vf + srcoff + d);
  bf16_t* vd = vT + ((size_t)h * DHEAD + d) * SEQ_ALL + soff + s;
  vd[0 * SEQ_ALL] = f2bf(xv.x);
  vd[1 * SEQ_ALL] = f2bf(xv.y);
  vd[2 * SEQ_ALL] = f2bf(xv.z);
  vd[3 * SEQ_ALL] = f2bf(xv.w);
}

// ------------------------------------------------------- Flash attention ----
// Block = 8 waves, ALL on the same head, 8 consecutive 16-row q-tiles.
// Per 32-key step the block stages K (32x128) and V (128x32) tiles into LDS
// with global_load_async_to_lds_b128 (ASYNCcnt), then every wave reads its
// fragments from LDS: 8x less global K/V traffic than per-wave streaming.
__global__ __launch_bounds__(256)
void k_flash_attn(const bf16_t* __restrict__ qb, const bf16_t* __restrict__ kb,
                  const bf16_t* __restrict__ vT, float* __restrict__ O) {
  __shared__ alignas(16) bf16_t sK[32 * 128];      // [key][d]
  __shared__ alignas(16) bf16_t sV[128 * 32];      // [d][key]
  __shared__ alignas(16) bf16_t pbuf[8][16 * 32];  // per-wave P staging (C->A layout)

  int tid  = threadIdx.x;
  int wave = tid >> 5;
  int lane = tid & 31;
  int h = blockIdx.y;
  int qbase = (blockIdx.x * 8 + wave) * 16;

  int n  = lane & 15;
  int mb = (lane >> 4) * 8;

  const bf16_t* qh = qb + ((size_t)h * SEQ_ALL + qbase) * DHEAD;
  const bf16_t* kh = kb + (size_t)h * SEQ_ALL * DHEAD;
  const bf16_t* vh = vT + (size_t)h * DHEAD * SEQ_ALL;

  bf16x16 qa[4];
  #pragma unroll
  for (int kk = 0; kk < 4; ++kk) qa[kk] = load_a_frag(qh + kk * 32, DHEAD);

  f32x8 o[8];
  #pragma unroll
  for (int nt = 0; nt < 8; ++nt) o[nt] = zero8();
  float mrow[8], lrow[8];
  #pragma unroll
  for (int r = 0; r < 8; ++r) { mrow[r] = -3.0e38f; lrow[r] = 0.0f; }

  for (int kv = 0; kv < SEQ_ALL; kv += 32) {
    // ---- async stage K/V tiles: 512 16B chunks each, 2 per thread ----
    {
      int c0 = tid, c1 = tid + 256;
      // K tile: same flat row-major order as global [key][d]
      async_ld_b128(sK + c0 * 8, kh + (size_t)kv * DHEAD + c0 * 8);
      async_ld_b128(sK + c1 * 8, kh + (size_t)kv * DHEAD + c1 * 8);
      // V tile: row d = c>>2, 8-elem group (c&3) of the 32 keys
      async_ld_b128(sV + c0 * 8, vh + (size_t)(c0 >> 2) * SEQ_ALL + kv + (c0 & 3) * 8);
      async_ld_b128(sV + c1 * 8, vh + (size_t)(c1 >> 2) * SEQ_ALL + kv + (c1 & 3) * 8);
    }
    wait_async0();
    __syncthreads();

    // scores: Q(16x128) x K^T(128x32) from sK ([key][d], ld=128)
    f32x8 s0 = zero8(), s1 = zero8();
    #pragma unroll
    for (int kk = 0; kk < 4; ++kk) {
      bf16x16 b0 = load_bT_frag(sK + kk * 32, DHEAD);
      bf16x16 b1 = load_bT_frag(sK + 16 * DHEAD + kk * 32, DHEAD);
      s0 = WMMA_BF16(qa[kk], b0, s0);
      s1 = WMMA_BF16(qa[kk], b1, s1);
    }
    // online softmax per row (rows live in 16-lane halves of each C VGPR)
    #pragma unroll
    for (int r = 0; r < 8; ++r) {
      float rm = fmaxf(s0[r], s1[r]);
      #pragma unroll
      for (int m = 1; m < 16; m <<= 1) rm = fmaxf(rm, __shfl_xor(rm, m, 32));
      float mn = fmaxf(mrow[r], rm);
      float alpha = __expf(mrow[r] - mn);
      float p0 = __expf(s0[r] - mn);
      float p1 = __expf(s1[r] - mn);
      float rs = p0 + p1;
      #pragma unroll
      for (int m = 1; m < 16; m <<= 1) rs += __shfl_xor(rs, m, 32);
      lrow[r] = lrow[r] * alpha + rs;
      mrow[r] = mn;
      #pragma unroll
      for (int nt = 0; nt < 8; ++nt) o[nt][r] *= alpha;
      pbuf[wave][(mb + r) * 32 + n]      = f2bf(p0);
      pbuf[wave][(mb + r) * 32 + 16 + n] = f2bf(p1);
    }
    // O += P(16x32) x V(32x128) from sV ([d][key], ld=32)
    bf16x16 pa = load_a_frag(&pbuf[wave][0], 32);
    #pragma unroll
    for (int nt = 0; nt < 8; ++nt) {
      bf16x16 bv = load_bT_frag(sV + (size_t)(nt * 16) * 32, 32);
      o[nt] = WMMA_BF16(pa, bv, o[nt]);
    }
    __syncthreads();   // protect sK/sV before next async stage
  }

  #pragma unroll
  for (int r = 0; r < 8; ++r) {
    float inv = 1.0f / lrow[r];
    #pragma unroll
    for (int nt = 0; nt < 8; ++nt) {
      O[(size_t)(qbase + mb + r) * DMODEL + h * DHEAD + nt * 16 + n] = o[nt][r] * inv;
    }
  }
}

// ------------------------------------------------------------- launcher -----
extern "C" void kernel_launch(void* const* d_in, const int* in_sizes, int n_in,
                              void* d_out, int out_size, void* d_ws, size_t ws_size,
                              hipStream_t stream) {
  const float* hs       = (const float*)d_in[0];
  const float* ehs      = (const float*)d_in[1];
  const float* img_rope = (const float*)d_in[2];
  const float* txt_rope = (const float*)d_in[3];
  const float* wq       = (const float*)d_in[4];
  const float* wk       = (const float*)d_in[5];
  const float* wv       = (const float*)d_in[6];
  const float* wq_txt   = (const float*)d_in[7];
  const float* wk_txt   = (const float*)d_in[8];
  const float* wv_txt   = (const float*)d_in[9];
  const float* gq       = (const float*)d_in[10];
  const float* gk       = (const float*)d_in[11];
  const float* gq_txt   = (const float*)d_in[12];
  const float* gk_txt   = (const float*)d_in[13];
  const float* w_out    = (const float*)d_in[14];
  const float* b_out    = (const float*)d_in[15];
  const float* w_add    = (const float*)d_in[16];
  const float* b_add    = (const float*)d_in[17];

  // workspace layout
  float* qi = (float*)d_ws;                      // 2048*3072 f32
  float* ki = qi + (size_t)SEQ_IMG * DMODEL;
  float* vi = ki + (size_t)SEQ_IMG * DMODEL;
  float* qt = vi + (size_t)SEQ_IMG * DMODEL;     // 512*3072 f32
  float* kt = qt + (size_t)SEQ_TXT * DMODEL;
  float* vt = kt + (size_t)SEQ_TXT * DMODEL;
  bf16_t* qb = (bf16_t*)(vt + (size_t)SEQ_TXT * DMODEL);   // 24*2560*128 bf16 each
  bf16_t* kb = qb + (size_t)HEADS * SEQ_ALL * DHEAD;
  bf16_t* vT = kb + (size_t)HEADS * SEQ_ALL * DHEAD;       // [h][d][s]
  float* O   = (float*)d_ws;   // 2560*3072 f32, reuses fp32 QKV region (phase-ordered)

  dim3 blk(256);

  // 1) QKV projections (bf16 WMMA GEMMs)
  k_gemm_bf16<<<dim3(DMODEL/128, SEQ_IMG/128), blk, 0, stream>>>(hs,  wq,     nullptr, qi, SEQ_IMG, DMODEL, DMODEL);
  k_gemm_bf16<<<dim3(DMODEL/128, SEQ_IMG/128), blk, 0, stream>>>(hs,  wk,     nullptr, ki, SEQ_IMG, DMODEL, DMODEL);
  k_gemm_bf16<<<dim3(DMODEL/128, SEQ_IMG/128), blk, 0, stream>>>(hs,  wv,     nullptr, vi, SEQ_IMG, DMODEL, DMODEL);
  k_gemm_bf16<<<dim3(DMODEL/128, SEQ_TXT/128), blk, 0, stream>>>(ehs, wq_txt, nullptr, qt, SEQ_TXT, DMODEL, DMODEL);
  k_gemm_bf16<<<dim3(DMODEL/128, SEQ_TXT/128), blk, 0, stream>>>(ehs, wk_txt, nullptr, kt, SEQ_TXT, DMODEL, DMODEL);
  k_gemm_bf16<<<dim3(DMODEL/128, SEQ_TXT/128), blk, 0, stream>>>(ehs, wv_txt, nullptr, vt, SEQ_TXT, DMODEL, DMODEL);

  // 2) RMSNorm + RoPE + pack (txt at s=0..511, img at 512..2559; V transposed)
  k_rmsrope<<<(SEQ_IMG * HEADS) / 8, blk, 0, stream>>>(qi, ki, vi, gq,     gk,     img_rope, qb, kb, vT, SEQ_TXT, SEQ_IMG);
  k_rmsrope<<<(SEQ_TXT * HEADS) / 8, blk, 0, stream>>>(qt, kt, vt, gq_txt, gk_txt, txt_rope, qb, kb, vT, 0,       SEQ_TXT);

  // 3) Flash attention (160 q-tiles, 24 heads; async K/V staging)
  k_flash_attn<<<dim3((SEQ_ALL/16) / 8, HEADS), blk, 0, stream>>>(qb, kb, vT, O);

  // 4) Output projections: d_out = [img_out (2048*3072) | txt_out (512*3072)]
  float* img_out = (float*)d_out;
  float* txt_out = img_out + (size_t)SEQ_IMG * DMODEL;
  k_gemm_bf16<<<dim3(DMODEL/128, SEQ_IMG/128), blk, 0, stream>>>(O + (size_t)SEQ_TXT * DMODEL, w_out, b_out, img_out, SEQ_IMG, DMODEL, DMODEL);
  k_gemm_bf16<<<dim3(DMODEL/128, SEQ_TXT/128), blk, 0, stream>>>(O,                            w_add, b_add, txt_out, SEQ_TXT, DMODEL, DMODEL);
}
